// Attention_fussion_41291815584276
// MI455X (gfx1250) — matrix operand order repaired and checked by reading, not compiled, not run on hardware
//
#include <hip/hip_runtime.h>

typedef __attribute__((ext_vector_type(16))) _Float16 v16h;
typedef __attribute__((ext_vector_type(8)))  _Float16 v8h;
typedef __attribute__((ext_vector_type(2)))  _Float16 v2h;
typedef __attribute__((ext_vector_type(8)))  float    v8f;

union F8 { v8f v; float f[8]; };
union H8 { v8h v; _Float16 h[8]; };
union H2 { v2h v; _Float16 h[2]; };

static constexpr int BATCH = 16;
static constexpr int CH    = 256;   // Cin == Cout == 256
static constexpr int HH    = 128;
static constexpr int WW    = 128;

__device__ __forceinline__ v16h cat8(v8h lo, v8h hi) {
  return __builtin_shufflevector(lo, hi, 0,1,2,3,4,5,6,7,8,9,10,11,12,13,14,15);
}
__device__ __forceinline__ v8f wmma_f16(v16h a, v16h b, v8f c) {
  // D = A(16x32 f16) * B(32x16 f16) + C(16x16 f32)
  return __builtin_amdgcn_wmma_f32_16x16x32_f16(false, a, false, b, (short)0, c, false, false);
}

// ---------------------------------------------------------------------------
// Kernel 1: QKV 3x3 conv as implicit GEMM.
// grid = (h=128, b=16, z=12) ; z -> set (q/k/v) * 4 N-tiles of 64 channels.
// Workgroup tile: M=128 (w positions) x N=64 (out channels), K=2304.
// LDS: xs[row 3][wp 130][ci 32] f16 (ci innermost -> A frags = 2x ds_load_b128)
//      wl[tap 9][co 64][ci 32]  f16 (ci innermost -> B frags = 2x ds_load_b128)
// Epilogue stages the 64x128 f16 tile in LDS for coalesced b128 stores.
// ---------------------------------------------------------------------------
__global__ __launch_bounds__(256, 2)
void conv_qkv_wmma(const float* __restrict__ x,
                   const float* __restrict__ Wq, const float* __restrict__ bq,
                   const float* __restrict__ Wk, const float* __restrict__ bk,
                   const float* __restrict__ Wv, const float* __restrict__ bv,
                   _Float16* __restrict__ qkv)
{
  const int h   = blockIdx.x;          // output row
  const int b   = blockIdx.y;          // batch
  const int z   = blockIdx.z;          // set*4 + ntile
  const int set = z >> 2;
  const int co0 = (z & 3) << 6;        // 64-channel N tile

  const float* Wset = (set == 0) ? Wq : (set == 1) ? Wk : Wv;
  const float* bset = (set == 0) ? bq : (set == 1) ? bk : bv;
  _Float16* outp = qkv + (size_t)set * ((size_t)BATCH * CH * HH * WW);

  __shared__ _Float16 xs[3 * 130 * 32];   // 24,960 B (reused as output stage)
  __shared__ _Float16 wl[9 * 64 * 32];    // 36,864 B

  const int tid  = threadIdx.x;
  const int lane = tid & 31;
  const int wid  = tid >> 5;   // 8 waves
  const int wm   = wid & 3;    // M group: w offset 32*wm
  const int wn   = wid >> 2;   // N group: co offset 32*wn
  const int hl   = lane >> 4;  // lane half (A and B K-base 0/16)
  const int l16  = lane & 15;

  // staging maps
  const int ci2 = (tid & 15) << 1;     // ci pair base 0..30
  const int w0  = (tid >> 4) << 3;     // 8-w group 0..120
  const int wco  = tid >> 2;           // weight: co 0..63
  const int wcib = (tid & 3) << 3;     // weight: ci base 0,8,16,24

  F8 acc[2][2] = {};

  for (int ci0 = 0; ci0 < CH; ci0 += 32) {
    __syncthreads();
    { // ---- stage x chunk: rows h-1..h+1, ci0..ci0+31, all w (+zero pads)
      #pragma unroll
      for (int r = 0; r < 3; ++r) {
        const int hh2 = h + r - 1;
        _Float16* dst = &xs[(r * 130 + (w0 + 1)) * 32 + ci2];
        if (hh2 >= 0 && hh2 < HH) {
          const float4* s0 =
              (const float4*)(x + (((size_t)b * CH + ci0 + ci2) * HH + hh2) * WW + w0);
          const float4* s1 =
              (const float4*)(x + (((size_t)b * CH + ci0 + ci2 + 1) * HH + hh2) * WW + w0);
          float4 a0 = s0[0], a1 = s0[1], b0 = s1[0], b1 = s1[1];
          const float fa[8] = {a0.x, a0.y, a0.z, a0.w, a1.x, a1.y, a1.z, a1.w};
          const float fb[8] = {b0.x, b0.y, b0.z, b0.w, b1.x, b1.y, b1.z, b1.w};
          #pragma unroll
          for (int j = 0; j < 8; ++j) {
            H2 t; t.h[0] = (_Float16)fa[j]; t.h[1] = (_Float16)fb[j];
            *(v2h*)&dst[j * 32] = t.v;
          }
        } else {
          #pragma unroll
          for (int j = 0; j < 8; ++j) *(v2h*)&dst[j * 32] = (v2h){};
        }
      }
      if (tid < 48) {                   // halo pads wp=0 and wp=129
        const int r = tid / 16, cp = (tid & 15) << 1;
        *(v2h*)&xs[(r * 130 + 0)   * 32 + cp] = (v2h){};
        *(v2h*)&xs[(r * 130 + 129) * 32 + cp] = (v2h){};
      }
    }
    { // ---- stage weights: wl[tap][co][ci] <- Wset[co0+co][ci0+ci][tap]
      #pragma unroll
      for (int tap = 0; tap < 9; ++tap) {
        H8 t;
        #pragma unroll
        for (int cj = 0; cj < 8; ++cj)
          t.h[cj] = (_Float16)Wset[((size_t)(co0 + wco) * CH + ci0 + wcib + cj) * 9 + tap];
        *(v8h*)&wl[(tap * 64 + wco) * 32 + wcib] = t.v;
      }
    }
    // prefetch next chunk of x rows while this chunk computes
    if (ci0 + 32 < CH) {
      #pragma unroll
      for (int r = 0; r < 3; ++r) {
        const int hh2 = h + r - 1;
        if (hh2 >= 0 && hh2 < HH)
          __builtin_prefetch(x + (((size_t)b * CH + ci0 + 32 + ci2) * HH + hh2) * WW + w0, 0, 1);
      }
    }
    __syncthreads();

    // ---- 9 taps x K=32 per chunk
    #pragma unroll
    for (int tap = 0; tap < 9; ++tap) {
      const int dh = tap / 3, dw = tap % 3;
      v16h afr[2], bfr[2];
      #pragma unroll
      for (int mi = 0; mi < 2; ++mi) {
        const int wp = (wm << 5) + (mi << 4) + l16 + dw;   // padded index (x[w+dw-1])
        const v8h* p = (const v8h*)&xs[(dh * 130 + wp) * 32 + hl * 16];
        afr[mi] = cat8(p[0], p[1]);     // A: K = base..base+15 contiguous
      }
      #pragma unroll
      for (int ni = 0; ni < 2; ++ni) {
        const int co = (wn << 5) + (ni << 4) + l16;
        const v8h* p = (const v8h*)&wl[(tap * 64 + co) * 32 + hl * 16];
        bfr[ni] = cat8(p[0], p[1]);     // B: K = base..base+15 contiguous
      }
      #pragma unroll
      for (int mi = 0; mi < 2; ++mi)
        #pragma unroll
        for (int ni = 0; ni < 2; ++ni)
          acc[mi][ni].v = wmma_f16(afr[mi], bfr[ni], acc[mi][ni].v);
    }
  }

  // ---- epilogue: +bias, pack f16, stage tile in LDS, coalesced b128 stores
  __syncthreads();                      // all waves done reading xs
  _Float16* ot = xs;                    // 64 co x 128 w f16 = 16 KB
  #pragma unroll
  for (int ni = 0; ni < 2; ++ni) {
    const int co_l = (wn << 5) + (ni << 4) + l16;
    const float bb = bset[co0 + co_l];
    #pragma unroll
    for (int mi = 0; mi < 2; ++mi) {
      const int wb = (wm << 5) + (mi << 4) + (hl << 3);   // M = r + 8*half
      H8 t;
      #pragma unroll
      for (int r = 0; r < 8; ++r)
        t.h[r] = (_Float16)(acc[mi][ni].f[r] + bb);
      *(v8h*)&ot[co_l * 128 + wb] = t.v;
    }
  }
  __syncthreads();
  const size_t obase = (((size_t)b * CH + co0) * HH + h) * WW;
  for (int i = tid; i < 1024; i += 256) {        // 64*128/8 v8h chunks
    const int co_l = i >> 4;
    const int c8   = (i & 15) << 3;
    *(v8h*)(outp + obase + (size_t)co_l * (HH * WW) + c8) =
        *(const v8h*)&ot[co_l * 128 + c8];
  }
}

// ---------------------------------------------------------------------------
// Kernel 2: per-(b,c) attention on 128x128 tiles.
// grid = 4096 workgroups (b*256+c), 256 threads = 8 waves, 16 rows per wave.
// Dynamic LDS (104,448 B): qs[128][136] (reused for P), ks[128][136],
// vt[136-strided transpose of v] so V serves as a WMMA B-matrix.
// q/k staged with gfx1250 async LDS loads (ASYNCcnt), v transposed manually.
// ---------------------------------------------------------------------------
__global__ __launch_bounds__(256, 1)
void attn_wmma(const _Float16* __restrict__ qkv, float* __restrict__ out)
{
  const int bc = blockIdx.x;
  const size_t setstr = (size_t)BATCH * CH * HH * WW;
  const size_t plane  = (size_t)bc * (HH * WW);
  const _Float16* qg = qkv + plane;
  const _Float16* kg = qkv + setstr + plane;
  const _Float16* vg = qkv + 2 * setstr + plane;

  extern __shared__ _Float16 smem[];
  _Float16* qs = smem;                 // 128*136, later reused for P
  _Float16* ks = smem + 128 * 136;
  _Float16* vt = smem + 2 * 128 * 136; // vt[w][g] = v[g][w]

  const int tid  = threadIdx.x;
  const int lane = tid & 31;
  const int wid  = tid >> 5;
  const int hl   = lane >> 4;
  const int l16  = lane & 15;

  // ---- stage q,k via async LDS loads; v transposed manually
  for (int i = tid; i < 128 * 16; i += 256) {
    const int row = i >> 4;
    const int c8  = (i & 15) << 3;
    const unsigned goff = (unsigned)((row * 128 + c8) * 2);          // bytes
    const unsigned lq = (unsigned)(uintptr_t)&qs[row * 136 + c8];    // LDS offset
    const unsigned lk = (unsigned)(uintptr_t)&ks[row * 136 + c8];
    asm volatile("global_load_async_to_lds_b128 %0, %1, %2 offset:0"
                 :: "v"(lq), "v"(goff), "s"(qg) : "memory");
    asm volatile("global_load_async_to_lds_b128 %0, %1, %2 offset:0"
                 :: "v"(lk), "v"(goff), "s"(kg) : "memory");
    H8 dv; dv.v = *(const v8h*)(vg + row * 128 + c8);
    #pragma unroll
    for (int j = 0; j < 8; ++j) vt[(c8 + j) * 136 + row] = dv.h[j];
  }
  asm volatile("s_wait_asynccnt 0x0" ::: "memory");
  __syncthreads();

  const int h0 = wid << 4;             // this wave's 16 query rows

  // ---- S = q @ k^T  (8 g-tiles x 4 K-steps)
  v16h aq[4];
  #pragma unroll
  for (int kc = 0; kc < 4; ++kc) {
    const v8h* p = (const v8h*)&qs[(h0 + l16) * 136 + (kc << 5) + hl * 16];
    aq[kc] = cat8(p[0], p[1]);
  }
  F8 s[8];
  #pragma unroll
  for (int t = 0; t < 8; ++t) {
    v8f a = {};
    #pragma unroll
    for (int kc = 0; kc < 4; ++kc) {
      const v8h* p = (const v8h*)&ks[((t << 4) + l16) * 136 + (kc << 5) + hl * 16];
      a = wmma_f16(aq[kc], cat8(p[0], p[1]), a);
    }
    s[t].v = a;
  }

  // ---- row softmax of S/sqrt(128) in C-fragment layout
  const float scale = 0.08838834764831845f;   // 1/sqrt(128)
  float inv_[8];
  #pragma unroll
  for (int r = 0; r < 8; ++r) {       // row M = r + 8*hl per lane-half
    float m = s[0].f[r];
    #pragma unroll
    for (int t = 1; t < 8; ++t) m = fmaxf(m, s[t].f[r]);
    #pragma unroll
    for (int off = 1; off < 16; off <<= 1) m = fmaxf(m, __shfl_xor(m, off, 32));
    float sum = 0.f;
    #pragma unroll
    for (int t = 0; t < 8; ++t) {
      float e = __expf((s[t].f[r] - m) * scale);
      s[t].f[r] = e;
      sum += e;
    }
    #pragma unroll
    for (int off = 1; off < 16; off <<= 1) sum += __shfl_xor(sum, off, 32);
    inv_[r] = 1.0f / sum;
  }

  // ---- P (f16) into reused q buffer; wave-local rows => no barrier needed
  #pragma unroll
  for (int t = 0; t < 8; ++t) {
    #pragma unroll
    for (int r = 0; r < 8; ++r)
      qs[(h0 + r + (hl << 3)) * 136 + (t << 4) + l16] = (_Float16)(s[t].f[r] * inv_[r]);
  }

  // ---- O = P @ V
  v16h ap[4];
  #pragma unroll
  for (int kc = 0; kc < 4; ++kc) {
    const v8h* p = (const v8h*)&qs[(h0 + l16) * 136 + (kc << 5) + hl * 16];
    ap[kc] = cat8(p[0], p[1]);
  }
  float* outp = out + (size_t)bc * (HH * WW);
  #pragma unroll
  for (int t = 0; t < 8; ++t) {
    v8f a = {};
    #pragma unroll
    for (int kc = 0; kc < 4; ++kc) {
      const v8h* p = (const v8h*)&vt[((t << 4) + l16) * 136 + (kc << 5) + hl * 16];
      a = wmma_f16(ap[kc], cat8(p[0], p[1]), a);
    }
    F8 o; o.v = a;
    #pragma unroll
    for (int r = 0; r < 8; ++r)
      outp[(h0 + r + (hl << 3)) * 128 + (t << 4) + l16] = o.f[r];
  }
}

extern "C" void kernel_launch(void* const* d_in, const int* in_sizes, int n_in,
                              void* d_out, int out_size, void* d_ws, size_t ws_size,
                              hipStream_t stream) {
  const float* x  = (const float*)d_in[0];
  const float* Wq = (const float*)d_in[1];
  const float* bq = (const float*)d_in[2];
  const float* Wk = (const float*)d_in[3];
  const float* bk = (const float*)d_in[4];
  const float* Wv = (const float*)d_in[5];
  const float* bv = (const float*)d_in[6];
  _Float16* qkv = (_Float16*)d_ws;   // 3 * 16*256*128*128 f16 = 402.7 MB
  float* out = (float*)d_out;

  const int attn_lds = 3 * 128 * 136 * (int)sizeof(_Float16);  // 104,448 B
  (void)hipFuncSetAttribute((const void*)attn_wmma,
                            hipFuncAttributeMaxDynamicSharedMemorySize, attn_lds);

  conv_qkv_wmma<<<dim3(HH, BATCH, 12), 256, 0, stream>>>(x, Wq, bq, Wk, bk, Wv, bv, qkv);
  attn_wmma<<<dim3(BATCH * CH), 256, attn_lds, stream>>>(qkv, out);
}
